// MoeMistralMLP_94489280671
// MI455X (gfx1250) — matrix operand order repaired and checked by reading, not compiled, not run on hardware
//
#include <hip/hip_runtime.h>

// ---------------------------------------------------------------------------
// MoE SwiGLU MLP for MI455X (gfx1250): bf16 WMMA (f32 accum) gather-GEMMs.
// B=4,S=2048 -> T=8192 tokens, H=1024, I=3584, E=8, TOP_K=2.
// Weights converted once per launch to bf16 [E][N][K] (176 MB, fits 192MB L2).
// ---------------------------------------------------------------------------

typedef __attribute__((ext_vector_type(8)))  __bf16 v8bf;
typedef __attribute__((ext_vector_type(16))) __bf16 v16bf;
typedef __attribute__((ext_vector_type(8)))  float  v8f;

#define WMMA_BF16(a, b, c) \
    __builtin_amdgcn_wmma_f32_16x16x32_bf16(false, (a), false, (b), (short)0, (c), false, false)

// ---------------------------------------------------------------- zero fill
__global__ void zero_f32_kernel(float* __restrict__ p, size_t n) {
    size_t i = (size_t)blockIdx.x * blockDim.x + threadIdx.x;
    if (i < n) p[i] = 0.0f;
}
__global__ void zero_i32_kernel(int* __restrict__ p, int n) {
    int i = blockIdx.x * blockDim.x + threadIdx.x;
    if (i < n) p[i] = 0;
}

// ---------------------------------------------------- f32 -> bf16 (straight)
__global__ void convert_kernel(const float* __restrict__ src, __bf16* __restrict__ dst, size_t n) {
    size_t i = (size_t)blockIdx.x * blockDim.x + threadIdx.x;
    if (i < n) dst[i] = (__bf16)src[i];
}

// -------------------------------- f32 [E][K][N] -> bf16 [E][N][K] (transpose)
__global__ void convert_transpose_kernel(const float* __restrict__ W, __bf16* __restrict__ Wt,
                                         int Kd, int Nd, int Ed) {
    size_t i = (size_t)blockIdx.x * blockDim.x + threadIdx.x;
    size_t total = (size_t)Ed * Kd * Nd;
    if (i >= total) return;
    int n = (int)(i % Nd);
    size_t t = i / Nd;
    int k = (int)(t % Kd);
    int e = (int)(t / Kd);
    Wt[((size_t)e * Nd + n) * Kd + k] = (__bf16)W[i];
}

// --------------------------------------------------------------- top-2 emit
__device__ inline void emit_top2(const float s[8], int* cnt, int* tok, float* w, int t, int T) {
    int i1 = 0; float v1 = s[0];
#pragma unroll
    for (int e = 1; e < 8; e++) if (s[e] > v1) { v1 = s[e]; i1 = e; }
    int i2 = (i1 == 0) ? 1 : 0; float v2 = s[i2];
#pragma unroll
    for (int e = 0; e < 8; e++) if (e != i1 && s[e] > v2) { v2 = s[e]; i2 = e; }
    // softmax over selected pair, renormalized: w1 = e^{v1}/(e^{v1}+e^{v2})
    float w1 = 1.0f / (1.0f + __expf(v2 - v1));
    float w2 = 1.0f - w1;
    int p1 = atomicAdd(&cnt[i1], 1); tok[i1 * T + p1] = t; w[i1 * T + p1] = w1;
    int p2 = atomicAdd(&cnt[i2], 1); tok[i2 * T + p2] = t; w[i2 * T + p2] = w2;
}

// ------------------------------------------- router: one wave per token
// Computes logits x_t . r[:,e] for (up to) two routers, builds per-expert
// compacted token lists. TIN = float (x) or __bf16 (h).
template <typename TIN, bool HASB>
__global__ void router_kernel(const TIN* __restrict__ xin, int ld,
                              const float* __restrict__ rA, const float* __restrict__ rB,
                              int* cntA, int* tokA, float* wA,
                              int* cntB, int* tokB, float* wB, int T) {
    int wid  = threadIdx.x >> 5;
    int lane = threadIdx.x & 31;
    int t = blockIdx.x * (blockDim.x >> 5) + wid;
    if (t >= T) return;
    float sA[8] = {0, 0, 0, 0, 0, 0, 0, 0};
    float sB[8] = {0, 0, 0, 0, 0, 0, 0, 0};
    const TIN* xp = xin + (size_t)t * ld;
    for (int k = lane; k < ld; k += 32) {
        float xv = (float)xp[k];
        const float* ra = rA + (size_t)k * 8;
#pragma unroll
        for (int e = 0; e < 8; e++) sA[e] += xv * ra[e];
        if (HASB) {
            const float* rb = rB + (size_t)k * 8;
#pragma unroll
            for (int e = 0; e < 8; e++) sB[e] += xv * rb[e];
        }
    }
#pragma unroll
    for (int e = 0; e < 8; e++) {
        for (int m = 16; m >= 1; m >>= 1) {
            sA[e] += __shfl_xor(sA[e], m, 32);
            if (HASB) sB[e] += __shfl_xor(sB[e], m, 32);
        }
    }
    if (lane == 0) {
        emit_top2(sA, cntA, tokA, wA, t, T);
        if (HASB) emit_top2(sB, cntB, tokB, wB, t, T);
    }
}

// ---------------------- A fragment: 16x32 bf16, two 16B chunks (ISA layout)
__device__ inline v16bf load_afrag(const __bf16* p) {
    v8bf lo = *(const v8bf*)p;          // K = base .. base+7
    v8bf hi = *(const v8bf*)(p + 16);   // K = base+16 .. base+23
    return __builtin_shufflevector(lo, hi, 0, 1, 2, 3, 4, 5, 6, 7,
                                           8, 9, 10, 11, 12, 13, 14, 15);
}

// ------------------------------------------------------- gather expert GEMM
// Out[t, :] += w[t,e] * (X[t, :] @ W_e)   for tokens routed to expert e.
// X: bf16 [T][K]; Wt: bf16 [E][N][K]; Out: f32 [T][N] (atomic accumulate).
// Block = 256 thr = 8 waves; tile = 128 tokens x 128 cols; K-step = 32.
__global__ __launch_bounds__(256) void moe_gemm_kernel(
    const __bf16* __restrict__ X, const __bf16* __restrict__ Wt,
    float* __restrict__ Out,
    const int* __restrict__ cnt, const int* __restrict__ tok,
    const float* __restrict__ wgt,
    int K, int N, int T, int tilesPerExpert) {
    const int e     = blockIdx.x / tilesPerExpert;
    const int mBase = (blockIdx.x % tilesPerExpert) * 128;
    const int count = cnt[e];
    if (mBase >= count) return;     // block-uniform early exit
    const int nBase = blockIdx.y * 128;

    __shared__ int   s_tok[128];
    __shared__ float s_w[128];
    const int tid = threadIdx.x;
    if (tid < 128) {
        int r = mBase + tid;
        int tv = -1; float wv = 0.0f;
        if (r < count) { tv = tok[e * T + r]; wv = wgt[e * T + r]; }
        s_tok[tid] = tv; s_w[tid] = wv;
    }
    __syncthreads();

    const int wid   = tid >> 5;
    const int lane  = tid & 31;
    const int waveM = (wid & 3) << 5;   // 4 waves along M: 32 rows each
    const int waveN = (wid >> 2) << 6;  // 2 waves along N: 64 cols each
    const int l15   = lane & 15;
    const int lhalf = lane >> 4;        // 0 | 1

    // Per-lane A row pointers (K-invariant): ISA A layout -> lane-half K off 8
    const int t0 = s_tok[waveM + l15];
    const int t1 = s_tok[waveM + 16 + l15];
    const __bf16* a0p = X + (size_t)(t0 < 0 ? 0 : t0) * K + (lhalf << 3);
    const __bf16* a1p = X + (size_t)(t1 < 0 ? 0 : t1) * K + (lhalf << 3);

    // Per-lane B column pointer: col = nBase+waveN+j*16+l15, lane-half K off 8
    const __bf16* wb = Wt + (size_t)e * N * K
                          + (size_t)(nBase + waveN + l15) * K + (lhalf << 3);
    const size_t cstride = (size_t)16 * K;  // 16 columns per N-fragment

    v8f acc[2][4] = {};
    for (int kk = 0; kk < K; kk += 32) {
        v16bf a[2], b[4];
        a[0] = load_afrag(a0p + kk);
        a[1] = load_afrag(a1p + kk);
        b[0] = load_afrag(wb + kk);
        b[1] = load_afrag(wb + cstride + kk);
        b[2] = load_afrag(wb + 2 * cstride + kk);
        b[3] = load_afrag(wb + 3 * cstride + kk);
#pragma unroll
        for (int i = 0; i < 2; i++)
#pragma unroll
            for (int j = 0; j < 4; j++)
                acc[i][j] = WMMA_BF16(a[i], b[j], acc[i][j]);
    }

    // Epilogue: C layout -> VGPR r: lanes 0-15 M=r, lanes 16-31 M=8+r; N=lane&15
    const int rAdd = lhalf << 3;  // 0 or 8
#pragma unroll
    for (int i = 0; i < 2; i++) {
        const int rb = waveM + (i << 4) + rAdd;
#pragma unroll
        for (int r = 0; r < 8; r++) {
            int row = rb + r;
            int tt = s_tok[row];
            if (tt < 0) continue;
            float wv = s_w[row];
            float* orow = Out + (size_t)tt * N + nBase + waveN + l15;
            atomicAdd(orow + 0,  wv * acc[i][0][r]);
            atomicAdd(orow + 16, wv * acc[i][1][r]);
            atomicAdd(orow + 32, wv * acc[i][2][r]);
            atomicAdd(orow + 48, wv * acc[i][3][r]);
        }
    }
}

// ----------------------------------------------------------- SwiGLU combine
__global__ void swiglu_kernel(const float* __restrict__ g, const float* __restrict__ u,
                              __bf16* __restrict__ h, size_t n) {
    size_t i = (size_t)blockIdx.x * blockDim.x + threadIdx.x;
    if (i >= n) return;
    float gv = g[i];
    float s  = gv / (1.0f + __expf(-gv));  // silu
    h[i] = (__bf16)(s * u[i]);
}

// ---------------------------------------------------------------------------
extern "C" void kernel_launch(void* const* d_in, const int* in_sizes, int n_in,
                              void* d_out, int out_size, void* d_ws, size_t ws_size,
                              hipStream_t stream) {
    (void)in_sizes; (void)n_in; (void)out_size; (void)ws_size;
    constexpr int H = 1024, I = 3584, E = 8;
    constexpr int T = 4 * 2048;  // B*S
    constexpr int TILES_T = T / 128;  // 64 (worst case: all tokens on one expert)

    const float* x      = (const float*)d_in[0];
    const float* gate_r = (const float*)d_in[1];
    const float* gate_w = (const float*)d_in[2];
    const float* up_r   = (const float*)d_in[3];
    const float* up_w   = (const float*)d_in[4];
    const float* down_r = (const float*)d_in[5];
    const float* down_w = (const float*)d_in[6];
    float* out = (float*)d_out;

    // ---- workspace carve-up (~487 MB) ----
    char* ws = (char*)d_ws;
    auto alloc = [&](size_t bytes) -> char* {
        char* p = ws;
        ws += (bytes + 255) & ~(size_t)255;
        return p;
    };
    __bf16* xb  = (__bf16*)alloc((size_t)T * H * 2);
    __bf16* gWt = (__bf16*)alloc((size_t)E * H * I * 2);
    __bf16* uWt = (__bf16*)alloc((size_t)E * H * I * 2);
    __bf16* dWt = (__bf16*)alloc((size_t)E * I * H * 2);
    float*  g   = (float*)alloc((size_t)T * I * 4);
    float*  u   = (float*)alloc((size_t)T * I * 4);
    __bf16* hb  = (__bf16*)alloc((size_t)T * I * 2);
    int*   cnt  = (int*)alloc(3 * E * sizeof(int));
    int*   tokG = (int*)alloc((size_t)E * T * 4);  float* wG = (float*)alloc((size_t)E * T * 4);
    int*   tokU = (int*)alloc((size_t)E * T * 4);  float* wU = (float*)alloc((size_t)E * T * 4);
    int*   tokD = (int*)alloc((size_t)E * T * 4);  float* wD = (float*)alloc((size_t)E * T * 4);
    int* cntG = cnt; int* cntU = cnt + E; int* cntD = cnt + 2 * E;

    auto blocks = [](size_t n, int bs) { return (unsigned)((n + bs - 1) / bs); };

    // 1) zero accumulators + counters (buffers are poisoned by harness)
    zero_f32_kernel<<<blocks((size_t)T * I, 256), 256, 0, stream>>>(g, (size_t)T * I);
    zero_f32_kernel<<<blocks((size_t)T * I, 256), 256, 0, stream>>>(u, (size_t)T * I);
    zero_f32_kernel<<<blocks((size_t)T * H, 256), 256, 0, stream>>>(out, (size_t)T * H);
    zero_i32_kernel<<<1, 32, 0, stream>>>(cnt, 3 * E);

    // 2) precision conversion: x -> bf16; weights -> bf16 [E][N][K] (transposed)
    convert_kernel<<<blocks((size_t)T * H, 256), 256, 0, stream>>>(x, xb, (size_t)T * H);
    convert_transpose_kernel<<<blocks((size_t)E * H * I, 256), 256, 0, stream>>>(gate_w, gWt, H, I, E);
    convert_transpose_kernel<<<blocks((size_t)E * H * I, 256), 256, 0, stream>>>(up_w, uWt, H, I, E);
    convert_transpose_kernel<<<blocks((size_t)E * I * H, 256), 256, 0, stream>>>(down_w, dWt, I, H, E);

    // 3) gate + up routing (both read x): one wave per token
    router_kernel<float, true><<<T / 8, 256, 0, stream>>>(
        x, H, gate_r, up_r, cntG, tokG, wG, cntU, tokU, wU, T);

    // 4) gate and up expert GEMMs: g/u[t,:] += w * (x @ W_e)
    {
        dim3 grid(E * TILES_T, I / 128);
        moe_gemm_kernel<<<grid, 256, 0, stream>>>(xb, gWt, g, cntG, tokG, wG, H, I, T, TILES_T);
        moe_gemm_kernel<<<grid, 256, 0, stream>>>(xb, uWt, u, cntU, tokU, wU, H, I, T, TILES_T);
    }

    // 5) h = silu(g) * u  (bf16 for the down GEMM)
    swiglu_kernel<<<blocks((size_t)T * I, 256), 256, 0, stream>>>(g, u, hb, (size_t)T * I);

    // 6) down routing on h
    router_kernel<__bf16, false><<<T / 8, 256, 0, stream>>>(
        hb, I, down_r, nullptr, cntD, tokD, wD, nullptr, nullptr, nullptr, T);

    // 7) down expert GEMM into d_out
    {
        dim3 grid(E * TILES_T, H / 128);
        moe_gemm_kernel<<<grid, 256, 0, stream>>>(hb, dWt, out, cntD, tokD, wD, I, H, T, TILES_T);
    }
}
